// CliffordFrameAttention_64939905515927
// MI455X (gfx1250) — matrix operand order repaired and checked by reading, not compiled, not run on hardware
//
#include <hip/hip_runtime.h>
#include <hip/hip_bf16.h>

#define BATCH   2
#define L_SEQ   2048
#define D       32
#define H       8
#define HIDDEN  256
#define CH      32          // keys per chunk (fills WMMA K-dim exactly)
#define NW      4           // waves per attention block
#define BROWS   (NW * 16)   // 64 query rows per block

#if __has_builtin(__builtin_amdgcn_global_load_async_to_lds_b128)
#define USE_ASYNC 1
#else
#define USE_ASYNC 0
#endif

typedef __attribute__((ext_vector_type(16))) __bf16 v16bf;
typedef __attribute__((ext_vector_type(2)))  __bf16 v2bf;
typedef __attribute__((ext_vector_type(8)))  float  v8f;

union U16 { v16bf v; unsigned int u[8]; };
union U1  { v2bf  v; unsigned int u; };

__device__ __forceinline__ unsigned short f2bf(float f) {
    unsigned int u = __float_as_uint(f);
    unsigned int r = u + 0x7FFFu + ((u >> 16) & 1u);   // round-to-nearest-even
    return (unsigned short)(r >> 16);
}

__device__ __forceinline__ unsigned int pack_bf16(float a, float b) {
#if __has_builtin(__builtin_amdgcn_cvt_pk_bf16_f32)
    auto r = __builtin_amdgcn_cvt_pk_bf16_f32(a, b);
    unsigned int u;
    __builtin_memcpy(&u, &r, 4);
    return u;
#else
    return (unsigned int)f2bf(a) | ((unsigned int)f2bf(b) << 16);
#endif
}

#if USE_ASYNC
typedef int v4i_t __attribute__((vector_size(16)));
typedef __attribute__((address_space(1))) v4i_t* g128_t;
typedef __attribute__((address_space(3))) v4i_t* l128_t;

__device__ __forceinline__ void async_b128(const void* g, void* l) {
    __builtin_amdgcn_global_load_async_to_lds_b128(
        (g128_t)(unsigned long long)(uintptr_t)g,
        (l128_t)(unsigned int)(uintptr_t)l,
        0, 0);
}
__device__ __forceinline__ void wait_async_le1() {
#if __has_builtin(__builtin_amdgcn_s_wait_asynccnt)
    __builtin_amdgcn_s_wait_asynccnt(1);
#else
    asm volatile("s_wait_asynccnt 0x1" ::: "memory");
#endif
}
__device__ __forceinline__ void wait_async_le0() {
#if __has_builtin(__builtin_amdgcn_s_wait_asynccnt)
    __builtin_amdgcn_s_wait_asynccnt(0);
#else
    asm volatile("s_wait_asynccnt 0x0" ::: "memory");
#endif
}
#endif

// ---------------------------------------------------------------------------
// Kernel 0a: bit-pack the mask, transposed.
// maskT[b][wr][col] bit r == mask[b][wr*32 + r][col]  (1 MB, L2-resident)
// ---------------------------------------------------------------------------
__global__ __launch_bounds__(256) void pack_mask_kernel(
    const unsigned char* __restrict__ mask,
    unsigned int* __restrict__ maskT)
{
    __shared__ unsigned int tile[32][64];        // 32 rows x 256 cols (bytes)
    const int tid = threadIdx.x;
    const int cg = blockIdx.x & 7;
    const int wr = (blockIdx.x >> 3) & 63;
    const int b  = blockIdx.x >> 9;

    const unsigned char* src = mask + ((size_t)b * L_SEQ + (size_t)wr * 32) * L_SEQ
                                    + (size_t)cg * 256;
#pragma unroll
    for (int t = 0; t < 8; ++t) {
        int idx = tid + 256 * t;
        int r = idx >> 6, c = idx & 63;
        tile[r][c] = ((const unsigned int*)(src + (size_t)r * L_SEQ))[c];
    }
    __syncthreads();
    const unsigned char* tb = (const unsigned char*)tile;
    unsigned int wword = 0;
#pragma unroll
    for (int r = 0; r < 32; ++r)
        wword |= (tb[r * 256 + tid] ? 1u : 0u) << r;
    maskT[((size_t)b * 64 + wr) * L_SEQ + (size_t)cg * 256 + tid] = wword;
}

// ---------------------------------------------------------------------------
// Kernel 0b: pack cayley (32x32x32 f32) into bf16 WMMA B-fragment layout.
// cayBuf[t][k][pp] packs p-pair (t*32+2pp, t*32+2pp+1) at column k.  64 KB.
// ---------------------------------------------------------------------------
__global__ __launch_bounds__(256) void pack_cayley_kernel(
    const float* __restrict__ cay, unsigned int* __restrict__ cayBuf)
{
    int id = blockIdx.x * 256 + threadIdx.x;     // < 16384
    int t  = id >> 9;
    int k  = (id >> 4) & 31;
    int pp = id & 15;
    size_t p0 = (size_t)t * 32 + 2 * pp;
    cayBuf[id] = pack_bf16(cay[p0 * 32 + k], cay[(p0 + 1) * 32 + k]);
}

// ---------------------------------------------------------------------------
// Kernel 1: Q/K/V projections (f32 + bf16 copies; signs/scale folded).
// ---------------------------------------------------------------------------
__global__ __launch_bounds__(256) void proj_kernel(
    const float* __restrict__ x,
    const float* __restrict__ Wq,
    const float* __restrict__ Wk,
    const float* __restrict__ Wv,
    const float* __restrict__ gsign,
    float* __restrict__ Qf, float* __restrict__ Kf,
    unsigned short* __restrict__ Qbf,
    unsigned short* __restrict__ Ksbf,
    unsigned short* __restrict__ Vbf)
{
    __shared__ float sx[D];
    const int tid = threadIdx.x;
    const int b = blockIdx.x / L_SEQ;
    const int l = blockIdx.x % L_SEQ;

    if (tid < D) sx[tid] = x[((size_t)b * L_SEQ + l) * D + tid];
    __syncthreads();

    const int h = tid >> 5;
    const int d = tid & 31;
    const float* wq = Wq + tid * D;
    const float* wk = Wk + tid * D;
    const float* wv = Wv + tid * D;

    float q = 0.f, k = 0.f, v = 0.f;
#pragma unroll
    for (int i = 0; i < D; ++i) {
        float xi = sx[i];
        q = fmaf(xi, wq[i], q);
        k = fmaf(xi, wk[i], k);
        v = fmaf(xi, wv[i], v);
    }
    size_t idx = ((size_t)(b * H + h) * L_SEQ + l) * D + d;
    Qf[idx] = q;
    Kf[idx] = k;
    Qbf[idx]  = f2bf(q * 0.17677669529663687f);   // fold 1/sqrt(32)
    Ksbf[idx] = f2bf(k * gsign[d]);
    Vbf[idx]  = f2bf(v);
}

// ---------------------------------------------------------------------------
// Kernel 2: fused flash attention (unchanged from round 5 — async K tiles,
// packed-bit mask, transposed V, all-lane online softmax, 4 WMMA per chunk).
// ---------------------------------------------------------------------------
__global__ __launch_bounds__(128) void attn_kernel(
    const unsigned short* __restrict__ Qbf,
    const unsigned short* __restrict__ Ksbf,
    const unsigned short* __restrict__ Vbf,
    const unsigned int*   __restrict__ maskT,
    float* __restrict__ Vagg)
{
    __shared__ __align__(16) unsigned int sKs[2][CH][16];
    __shared__ __align__(16) unsigned int sVt[D][CH / 2];
    __shared__ float        sS[NW][16][32];
    __shared__ unsigned int sP[NW][16][16];
    __shared__ unsigned int sQt[NW][16][16];
    __shared__ float sM[NW][16], sSum[NW][16], sCorr[NW][16];

    const int tid = threadIdx.x;
    const int w   = tid >> 5;
    const int ln  = tid & 31;
    const int hiL = ln >> 4;
    const int lo  = ln & 15;

    const int nRB = L_SEQ / BROWS;
    const int rb  = blockIdx.x % nRB;
    const int h   = (blockIdx.x / nRB) % H;
    const int b   = blockIdx.x / (nRB * H);
    const int row0 = rb * BROWS;

    const size_t headOff = (size_t)(b * H + h) * L_SEQ * D;
    const unsigned short* Kbase = Ksbf + headOff;
    const unsigned short* Vbase = Vbf + headOff;

    const int wrW     = (row0 + w * 16 + 8 * hiL) >> 5;
    const int bitBase = (w * 16 + 8 * hiL) & 31;
    const unsigned int* MTbase = maskT + ((size_t)b * (L_SEQ / 32) + wrW) * L_SEQ;

    const int kkA = tid >> 2;
    const int kcA = tid & 3;
    const int NCHUNK = L_SEQ / CH;

#if USE_ASYNC
    async_b128(Kbase + (size_t)kkA * D + kcA * 8, &sKs[0][kkA][kcA * 4]);
#else
    unsigned int rK[4];
#pragma unroll
    for (int t = 0; t < 4; ++t) {
        int idx = tid + 128 * t;
        rK[t] = ((const unsigned int*)(Kbase + (size_t)(idx >> 4) * D))[idx & 15];
    }
#endif
    uint4 rV = ((const uint4*)(Vbase + (size_t)kkA * D))[kcA];
    unsigned int mw0 = MTbase[lo];
    unsigned int mw1 = MTbase[16 + lo];

    for (int i = tid; i < BROWS * 16; i += 128) {
        int r = i >> 4, c = i & 15;
        sQt[r >> 4][r & 15][c] =
            ((const unsigned int*)(Qbf + headOff + (size_t)(row0 + r) * D))[c];
    }
    if (ln < 16) { sM[w][lo] = -__builtin_inff(); sSum[w][lo] = 0.f; }
    __syncthreads();

    U16 aQ;
#pragma unroll
    for (int vv = 0; vv < 8; ++vv) {
        int k0 = 2 * vv + hiL * 16;
        aQ.u[vv] = sQt[w][lo][k0 >> 1];
    }

    v8f o0 = {0.f,0.f,0.f,0.f,0.f,0.f,0.f,0.f};
    v8f o1 = {0.f,0.f,0.f,0.f,0.f,0.f,0.f,0.f};

    for (int cc = 0; cc < NCHUNK; ++cc) {
        const int buf = cc & 1;
        __syncthreads();

        unsigned short* sVtUs = (unsigned short*)sVt;
        {
            unsigned int rv4[4] = {rV.x, rV.y, rV.z, rV.w};
#pragma unroll
            for (int u = 0; u < 4; ++u) {
                int d0 = kcA * 8 + 2 * u;
                sVtUs[d0 * CH + kkA]       = (unsigned short)(rv4[u] & 0xFFFFu);
                sVtUs[(d0 + 1) * CH + kkA] = (unsigned short)(rv4[u] >> 16);
            }
        }
#if !USE_ASYNC
#pragma unroll
        for (int t = 0; t < 4; ++t) {
            int idx = tid + 128 * t;
            sKs[buf][idx >> 4][idx & 15] = rK[t];
        }
#endif
        unsigned int nmw0 = 0, nmw1 = 0;
        if (cc + 1 < NCHUNK) {
            const size_t c0n = (size_t)(cc + 1) * CH;
#if USE_ASYNC
            async_b128(Kbase + c0n * D + (size_t)kkA * D + kcA * 8,
                       &sKs[buf ^ 1][kkA][kcA * 4]);
#else
#pragma unroll
            for (int t = 0; t < 4; ++t) {
                int idx = tid + 128 * t;
                rK[t] = ((const unsigned int*)(Kbase + c0n * D + (size_t)(idx >> 4) * D))[idx & 15];
            }
#endif
            rV   = ((const uint4*)(Vbase + c0n * D + (size_t)kkA * D))[kcA];
            nmw0 = MTbase[c0n + lo];
            nmw1 = MTbase[c0n + 16 + lo];
        }
#if USE_ASYNC
        if (cc + 1 < NCHUNK) wait_async_le1();
        else                 wait_async_le0();
#endif
        __syncthreads();

#pragma unroll
        for (int half = 0; half < 2; ++half) {
            U16 bK;
#pragma unroll
            for (int vv = 0; vv < 8; ++vv)
                bK.u[vv] = sKs[buf][half * 16 + lo][vv + 8 * hiL];
            v8f zc = {0.f,0.f,0.f,0.f,0.f,0.f,0.f,0.f};
            v8f s = __builtin_amdgcn_wmma_f32_16x16x32_bf16(
                        false, aQ.v, false, bK.v, (short)0, zc, false, false);
            const unsigned int mw = half ? mw1 : mw0;
#pragma unroll
            for (int e = 0; e < 8; ++e) {
                int cl = half * 16 + lo;
                bool mv = (mw >> (bitBase + e)) & 1u;
                sS[w][e + 8 * hiL][cl] = mv ? s[e] : -1.0e9f;
            }
        }
        __syncthreads();

        {
            float mold = sM[w][lo];
            float cmax = -__builtin_inff();
#pragma unroll
            for (int j = 0; j < 16; ++j)
                cmax = fmaxf(cmax, sS[w][lo][hiL * 16 + j]);
            cmax = fmaxf(cmax, __shfl_xor(cmax, 16));
            float mnew = fmaxf(mold, cmax);
            float corr = __expf(mold - mnew);
            float ssum = 0.f;
#pragma unroll
            for (int j = 0; j < 16; j += 2) {
                float p0 = __expf(sS[w][lo][hiL * 16 + j]     - mnew);
                float p1 = __expf(sS[w][lo][hiL * 16 + j + 1] - mnew);
                ssum += p0 + p1;
                sP[w][lo][hiL * 8 + (j >> 1)] = pack_bf16(p0, p1);
            }
            ssum += __shfl_xor(ssum, 16);
            if (ln < 16) {
                sSum[w][lo] = sSum[w][lo] * corr + ssum;
                sM[w][lo]   = mnew;
                sCorr[w][lo] = corr;
            }
        }
        __syncthreads();

#pragma unroll
        for (int e = 0; e < 8; ++e) {
            float f = sCorr[w][e + 8 * hiL];
            o0[e] *= f;
            o1[e] *= f;
        }

        U16 aP;
#pragma unroll
        for (int vv = 0; vv < 8; ++vv) {
            int k0 = 2 * vv + hiL * 16;
            aP.u[vv] = sP[w][lo][k0 >> 1];
        }
        U16 bV0, bV1;
#pragma unroll
        for (int vv = 0; vv < 8; ++vv) {
            bV0.u[vv] = sVt[lo]     [vv + 8 * hiL];
            bV1.u[vv] = sVt[lo + 16][vv + 8 * hiL];
        }
        o0 = __builtin_amdgcn_wmma_f32_16x16x32_bf16(
                 false, aP.v, false, bV0.v, (short)0, o0, false, false);
        o1 = __builtin_amdgcn_wmma_f32_16x16x32_bf16(
                 false, aP.v, false, bV1.v, (short)0, o1, false, false);

        mw0 = nmw0;
        mw1 = nmw1;
    }
    __syncthreads();

#pragma unroll
    for (int e = 0; e < 8; ++e) {
        int r = e + 8 * hiL;
        float inv = 1.f / sSum[w][r];
        size_t base = headOff + (size_t)(row0 + w * 16 + r) * D;
        Vagg[base + lo]      = o0[e] * inv;
        Vagg[base + lo + 16] = o1[e] * inv;
    }
}

// ---------------------------------------------------------------------------
// Kernel 3 (WMMA rewrite): Clifford geometric products as a per-token GEMM.
//   OUT[m,k] = sum_p COEF[m,p] * cayley[p,k],  m = 2h+e (16 rows), p = (i,j).
// A 32-wide p-chunk has fixed i=t, so the A-fragment is q_m(t) * Z_m[j]:
// 8 packed-bf16 muls per lane — no COEF materialization. cayley bf16 lives in
// LDS in B-fragment layout (64 KB; CDNA5's 320 KB/WGP), shared by 4 tokens.
// One wave per token: 64 WMMAs/token replace ~64K scalar FMAs + 1M LDS reads.
//   y[h*32+k] = OUT[2h,k] + Vagg + 0.25*OUT[2h+1,k];  out = rv * (y @ Wo^T).
// ---------------------------------------------------------------------------
__global__ __launch_bounds__(128) void clifford_out_kernel(
    const float* __restrict__ Qf,
    const float* __restrict__ Kf,
    const float* __restrict__ Vagg,
    const unsigned char* __restrict__ mask,
    const float* __restrict__ Wo,
    const unsigned int* __restrict__ cayBuf,
    float* __restrict__ out)
{
    extern __shared__ unsigned int cayB[];       // [32][32][16] uints, 64 KB
    __shared__ float        sQ[4][8][32];        // per-wave Q rows (f32)
    __shared__ float        sV[4][8][32];        // per-wave V_agg rows (f32)
    __shared__ unsigned int sZ[4][16][16];       // per-wave Z bf16 pairs [m][j/2]
    __shared__ float        sy[4][HIDDEN];

    const int tid = threadIdx.x;
    const int w   = tid >> 5;
    const int ln  = tid & 31;
    const int hiL = ln >> 4;
    const int lo  = ln & 15;

    const int tok = blockIdx.x * 4 + w;
    const int b = tok / L_SEQ;
    const int l = tok % L_SEQ;

    // ---- cayley fragments -> LDS (block-cooperative, b128 copies) ----
    {
        const uint4* src = (const uint4*)cayBuf;
        uint4* dst = (uint4*)cayB;
        for (int i = tid; i < 4096; i += 128) dst[i] = src[i];
    }

    // ---- per-wave staging: Q/Vagg f32, Z rows (V_agg & K) as bf16 pairs ----
    for (int i = ln; i < 256; i += 32) {
        int h = i >> 5, d = i & 31;
        size_t idx = ((size_t)(b * H + h) * L_SEQ + l) * D + d;
        sQ[w][h][d] = Qf[idx];
        sV[w][h][d] = Vagg[idx];
    }
    for (int i = ln; i < 256; i += 32) {
        int m = i >> 4, pp = i & 15, h = m >> 1;
        size_t idx = ((size_t)(b * H + h) * L_SEQ + l) * D + 2 * pp;
        if (m & 1) sZ[w][m][pp] = pack_bf16(Kf[idx],   Kf[idx + 1]);
        else       sZ[w][m][pp] = pack_bf16(Vagg[idx], Vagg[idx + 1]);
    }

    // ---- row_valid = any(mask[b,l,:]) via b128 scans + wave vote ----
    unsigned int any = 0;
    {
        const uint4* mrow = (const uint4*)(mask + ((size_t)b * L_SEQ + l) * L_SEQ);
        for (int i = ln; i < 128; i += 32) {
            uint4 mv = mrow[i];
            any |= mv.x | mv.y | mv.z | mv.w;
        }
    }
    const float rv = __any(any != 0) ? 1.f : 0.f;
    __syncthreads();

    // ---- 32 chunk-steps of the 16x32x1024 GEMM ----
    v8f o0 = {0.f,0.f,0.f,0.f,0.f,0.f,0.f,0.f};
    v8f o1 = {0.f,0.f,0.f,0.f,0.f,0.f,0.f,0.f};
    for (int t = 0; t < 32; ++t) {
        float qf = sQ[w][lo >> 1][t];            // scalar for A row m = lo
        U1 qd;
        __bf16 qb = (__bf16)qf;
        qd.v = (v2bf){qb, qb};
        U16 aC;
#pragma unroll
        for (int vv = 0; vv < 8; ++vv) {
            int j0 = 2 * vv + hiL * 16;
            U1 z; z.u = sZ[w][lo][j0 >> 1];
            U1 r; r.v = z.v * qd.v;              // v_pk_mul_bf16
            aC.u[vv] = r.u;
        }
        U16 bB0, bB1;
        const unsigned int* cb = cayB + t * 512;
#pragma unroll
        for (int vv = 0; vv < 8; ++vv) {
            bB0.u[vv] = cb[lo * 16        + vv + 8 * hiL];
            bB1.u[vv] = cb[(lo + 16) * 16 + vv + 8 * hiL];
        }
        o0 = __builtin_amdgcn_wmma_f32_16x16x32_bf16(
                 false, aC.v, false, bB0.v, (short)0, o0, false, false);
        o1 = __builtin_amdgcn_wmma_f32_16x16x32_bf16(
                 false, aC.v, false, bB1.v, (short)0, o1, false, false);
    }

    // ---- combine: y[h*32+k] = OUT[2h,k] + Vagg + 0.25*OUT[2h+1,k] ----
#pragma unroll
    for (int q = 0; q < 4; ++q) {
        int h = hiL * 4 + q;                     // C rows r=2q,2q+1 -> m=2h,2h+1
        sy[w][h * 32 + lo]      = o0[2 * q] + 0.25f * o0[2 * q + 1] + sV[w][h][lo];
        sy[w][h * 32 + lo + 16] = o1[2 * q] + 0.25f * o1[2 * q + 1] + sV[w][h][lo + 16];
    }
    __syncthreads();

    // ---- output projection: lane d = ln over 256-wide dot ----
    float acc = 0.f;
    const float* wo = Wo + ln * HIDDEN;
    for (int c = 0; c < HIDDEN; ++c)
        acc = fmaf(sy[w][c], wo[c], acc);
    out[((size_t)b * L_SEQ + l) * D + ln] = acc * rv;
}

// ---------------------------------------------------------------------------
extern "C" void kernel_launch(void* const* d_in, const int* in_sizes, int n_in,
                              void* d_out, int out_size, void* d_ws, size_t ws_size,
                              hipStream_t stream) {
    const float*         x      = (const float*)d_in[0];
    const unsigned char* mask   = (const unsigned char*)d_in[1];  // bool array
    const float*         Wq     = (const float*)d_in[2];
    const float*         Wk     = (const float*)d_in[3];
    const float*         Wv     = (const float*)d_in[4];
    const float*         Wo     = (const float*)d_in[5];
    const float*         cayley = (const float*)d_in[6];
    const float*         gsign  = (const float*)d_in[7];
    float*               out    = (float*)d_out;
    (void)in_sizes; (void)n_in; (void)out_size; (void)ws_size;

    const size_t NQ = (size_t)BATCH * H * L_SEQ * D;   // 1,048,576
    char* ws = (char*)d_ws;
    float* Qf   = (float*)ws;          ws += NQ * 4;
    float* Kf   = (float*)ws;          ws += NQ * 4;
    float* Vagg = (float*)ws;          ws += NQ * 4;
    unsigned short* Qbf  = (unsigned short*)ws; ws += NQ * 2;
    unsigned short* Ksbf = (unsigned short*)ws; ws += NQ * 2;
    unsigned short* Vbf  = (unsigned short*)ws; ws += NQ * 2;
    unsigned int* maskT  = (unsigned int*)ws;   ws += (size_t)BATCH * (L_SEQ / 32) * L_SEQ * 4;
    unsigned int* cayBuf = (unsigned int*)ws;   ws += 16384 * 4;

    pack_mask_kernel<<<BATCH * 64 * 8, 256, 0, stream>>>(mask, maskT);
    pack_cayley_kernel<<<64, 256, 0, stream>>>(cayley, cayBuf);

    proj_kernel<<<BATCH * L_SEQ, 256, 0, stream>>>(
        x, Wq, Wk, Wv, gsign, Qf, Kf, Qbf, Ksbf, Vbf);

    attn_kernel<<<BATCH * H * (L_SEQ / BROWS), 128, 0, stream>>>(
        Qbf, Ksbf, Vbf, maskT, Vagg);

    clifford_out_kernel<<<BATCH * L_SEQ / 4, 128, 16384 * 4, stream>>>(
        Qf, Kf, Vagg, mask, Wo, cayBuf, out);
}
